// DynamicShortConvolution_24043226923602
// MI455X (gfx1250) — matrix operand order repaired and checked by reading, not compiled, not run on hardware
//
#include <hip/hip_runtime.h>
#include <hip/hip_bf16.h>

// Shapes (fixed by the reference)
#define TT 16384
#define DD 2048
#define WW 4
#define GH 128
#define BB 8
#define CW (DD * WW)   // 8192 kern columns

typedef __attribute__((ext_vector_type(16))) __bf16 v16bf;
typedef __attribute__((ext_vector_type(8)))  float  v8f;

__device__ __forceinline__ float silu_f(float v) {
    // v * rcp(1 + e^-v): v_exp_f32 + v_rcp_f32, no div-scale fixup chains
    return v * __builtin_amdgcn_rcpf(1.0f + __expf(-v));
}

// ---------------------------------------------------------------------------
// Pre-pack a (K x N) f32 matrix into bf16 WMMA B-operand fragments:
// block (ct, kc) holds a 32x16 K-chunk; lane l, element j <-
//   src[(kc*32 + (j&7) + 8*(l>>4) + 16*(j>>3)) * N + ct*16 + (l&15)]
// One packed fragment per lane is then a contiguous 32-byte load in the GEMM.
// ---------------------------------------------------------------------------
__global__ void pack_b_kernel(const float* __restrict__ src,
                              __bf16* __restrict__ dst,
                              int K, int N) {
    const int KC = K >> 5;
    int e = blockIdx.x * 256 + threadIdx.x;
    if (e >= K * N) return;
    const int j  = e & 15;
    const int l  = (e >> 4) & 31;
    const int kc = (e >> 9) % KC;
    const int ct = e / (KC << 9);
    const int k  = kc * 32 + (j & 7) + ((l >> 4) << 3) + ((j >> 3) << 4);
    const int n  = ct * 16 + (l & 15);
    dst[e] = (__bf16)src[(size_t)k * N + n];
}

// ---------------------------------------------------------------------------
// Phase 1: H = silu(G @ W1); store H directly as packed bf16 A-fragments (Hp)
// so phase 2 needs zero conversions. One wave -> one 16x16 tile, K=2048.
// ---------------------------------------------------------------------------
__global__ void gen_h_kernel(const float*  __restrict__ G,
                             const __bf16* __restrict__ W1p,
                             __bf16*       __restrict__ Hp) {
    const int wave = threadIdx.x >> 5;
    const int lane = threadIdx.x & 31;
    const int tile = blockIdx.x * 4 + wave;   // 8192 tiles
    const int tt = tile >> 3;                 // t-tile 0..1023
    const int ct = tile & 7;                  // col-tile 0..7
    const int t0 = tt * 16, c0 = ct * 16;

    const int row = lane & 15;
    const int kbA = (lane >> 4) * 8;

    v8f acc = {};
    for (int kc = 0; kc < GH * (DD / GH) / 32; ++kc) {   // 64 K-chunks of 32
        const float* grow = G + (size_t)(t0 + row) * DD + kc * 32 + kbA;
        const float4 f0 = *(const float4*)(grow);
        const float4 f1 = *(const float4*)(grow + 4);
        const float4 f2 = *(const float4*)(grow + 16);
        const float4 f3 = *(const float4*)(grow + 20);
        v16bf a;
        a[0]=(__bf16)f0.x;  a[1]=(__bf16)f0.y;  a[2]=(__bf16)f0.z;  a[3]=(__bf16)f0.w;
        a[4]=(__bf16)f1.x;  a[5]=(__bf16)f1.y;  a[6]=(__bf16)f1.z;  a[7]=(__bf16)f1.w;
        a[8]=(__bf16)f2.x;  a[9]=(__bf16)f2.y;  a[10]=(__bf16)f2.z; a[11]=(__bf16)f2.w;
        a[12]=(__bf16)f3.x; a[13]=(__bf16)f3.y; a[14]=(__bf16)f3.z; a[15]=(__bf16)f3.w;
        const v16bf b = *(const v16bf*)(W1p + (((size_t)ct * 64 + kc) * 32 + lane) * 16);
        acc = __builtin_amdgcn_wmma_f32_16x16x32_bf16(
            false, a, false, b, (short)0, acc, false, false);
    }

    // Scatter silu(acc) into the packed A layout of Hp:
    // value (m = j + 8*(lane>=16), g = c0 + lane&15) ->
    //   chunk kc = g>>5, lane la = m + 16*((g>>3)&1 of low band), elem jdst
    const int g    = c0 + (lane & 15);
    const int kk   = g & 31;
    const int kcD  = g >> 5;
    const int jdst = (kk & 7) + ((kk >= 16) ? 8 : 0);
    const int lb   = ((kk >> 3) & 1) << 4;
#pragma unroll
    for (int j = 0; j < 8; ++j) {
        const int m = j + ((lane >> 4) << 3);
        Hp[(((size_t)tt * 4 + kcD) * 32 + (m + lb)) * 16 + jdst] =
            (__bf16)silu_f(acc[j]);
    }
}

// ---------------------------------------------------------------------------
// Phase 2 (fused): kern tile = Hp(16x128) @ W2p(128x16) + b2; depthwise conv
// over W taps + silu, all in registers. Columns map to (d = col/4, w = col%4).
// ---------------------------------------------------------------------------
__global__ void conv_fused_kernel(const float*  __restrict__ x,
                                  const __bf16* __restrict__ Hp,
                                  const __bf16* __restrict__ W2p,
                                  const float*  __restrict__ b2,
                                  const int*    __restrict__ cu_seqlens,
                                  const int*    __restrict__ seq_idx,
                                  const int*    __restrict__ cache_indices,
                                  const unsigned char* __restrict__ has_init,
                                  const float*  __restrict__ cache,
                                  float*        __restrict__ out) {
    const int wave = threadIdx.x >> 5;
    const int lane = threadIdx.x & 31;
    const int tt = blockIdx.x;                 // 0..1023
    const int ct = blockIdx.y * 8 + wave;      // 0..511
    const int t0 = tt * 16, c0 = ct * 16;

    v8f acc = {};
#pragma unroll
    for (int kc = 0; kc < 4; ++kc) {
        const v16bf a = *(const v16bf*)(Hp  + (((size_t)tt * 4 + kc) * 32 + lane) * 16);
        const v16bf b = *(const v16bf*)(W2p + (((size_t)ct * 4 + kc) * 32 + lane) * 16);
        acc = __builtin_amdgcn_wmma_f32_16x16x32_bf16(
            false, a, false, b, (short)0, acc, false, false);
    }

    // Column decode: col = c0 + n ; d = col>>2 ; w = col&3 ; o = 3 - w
    const int n   = lane & 15;
    const int col = c0 + n;
    const int d   = col >> 2;
    const int o   = 3 - (col & 3);
    const float bias = b2[col];

#pragma unroll
    for (int j = 0; j < 8; ++j) {
        const int m   = j + ((lane >> 4) << 3);
        const int t   = t0 + m;
        const int si  = seq_idx[t];
        const int pos = t - cu_seqlens[si];
        const int src = pos - o;
        float ctx;
        if (src >= 0) {
            ctx = x[(size_t)(t - o) * DD + d];
        } else if (has_init[si]) {
            int c = WW + src; c = c < 0 ? 0 : (c > WW - 1 ? WW - 1 : c);
            ctx = cache[((size_t)cache_indices[si] * DD + d) * WW + c];
        } else {
            ctx = 0.0f;
        }
        float v = (acc[j] + bias) * ctx;
        // reduce the 4 taps (adjacent lanes share d, differ in w)
        v += __shfl_xor(v, 1, 32);
        v += __shfl_xor(v, 2, 32);
        if ((lane & 3) == 0) {
            out[(size_t)t * DD + d] = silu_f(v);
        }
    }
}

// ---------------------------------------------------------------------------
// Phase 3: cache_new = cache; then overwrite cols 1..W-1 per reference
// ---------------------------------------------------------------------------
__global__ void cache_copy_kernel(const float* __restrict__ cache,
                                  float* __restrict__ cache_out) {
    int i = blockIdx.x * 256 + threadIdx.x;
    if (i < BB * DD * WW) cache_out[i] = cache[i];
}

__global__ void cache_update_kernel(const float* __restrict__ x,
                                    const float* __restrict__ cache,
                                    const int*   __restrict__ cu_seqlens,
                                    const int*   __restrict__ cache_indices,
                                    const unsigned char* __restrict__ has_init,
                                    float* __restrict__ cache_out) {
    int i = blockIdx.x * 256 + threadIdx.x;
    if (i >= BB * DD) return;
    const int b = i / DD, d = i % DD;
    const int ci    = cache_indices[b];
    const int start = cu_seqlens[b];
    const int L     = cu_seqlens[b + 1] - start;
    const bool hi   = has_init[b] != 0;
#pragma unroll
    for (int k = 0; k < WW - 1; ++k) {
        const int rel = L - (WW - 1) + k;
        float v;
        if (rel >= 0) {
            int tok = start + rel;
            tok = tok < 0 ? 0 : (tok > TT - 1 ? TT - 1 : tok);
            v = x[(size_t)tok * DD + d];
        } else if (hi) {
            int c = WW + rel; c = c < 0 ? 0 : (c > WW - 1 ? WW - 1 : c);
            v = cache[((size_t)ci * DD + d) * WW + c];
        } else {
            v = cache[((size_t)ci * DD + d) * WW + 1 + k];
        }
        cache_out[((size_t)ci * DD + d) * WW + 1 + k] = v;
    }
}

extern "C" void kernel_launch(void* const* d_in, const int* in_sizes, int n_in,
                              void* d_out, int out_size, void* d_ws, size_t ws_size,
                              hipStream_t stream) {
    const float* x     = (const float*)d_in[0];
    const float* G     = (const float*)d_in[1];
    const float* W1    = (const float*)d_in[2];
    const float* W2    = (const float*)d_in[3];
    const float* b2    = (const float*)d_in[4];
    const float* cache = (const float*)d_in[5];
    const int* cu      = (const int*)d_in[6];
    const int* sidx    = (const int*)d_in[7];
    const int* cidx    = (const int*)d_in[8];
    const unsigned char* hinit = (const unsigned char*)d_in[9];

    float* out       = (float*)d_out;
    float* cache_out = out + (size_t)TT * DD;

    // Workspace layout (bf16): Hp 4MB | W1p 0.5MB | W2p 2MB
    __bf16* Hp  = (__bf16*)d_ws;
    __bf16* W1p = (__bf16*)((char*)d_ws + (4u << 20));
    __bf16* W2p = (__bf16*)((char*)d_ws + (4u << 20) + (1u << 19));

    // One-time weight pre-pack into WMMA B-operand fragment layout
    pack_b_kernel<<<(DD * GH + 255) / 256, 256, 0, stream>>>(W1, W1p, DD, GH);
    pack_b_kernel<<<(GH * CW + 255) / 256, 256, 0, stream>>>(W2, W2p, GH, CW);

    // Phase 1: 8192 tiles, 4 waves/block -> 2048 blocks of 128 threads
    gen_h_kernel<<<2048, 128, 0, stream>>>(G, W1p, Hp);

    // Phase 2: grid (T/16, 512/8), 8 waves/block of 256 threads
    conv_fused_kernel<<<dim3(1024, 64), 256, 0, stream>>>(
        x, Hp, W2p, b2, cu, sidx, cidx, hinit, cache, out);

    // Phase 3: cache_new
    cache_copy_kernel<<<(BB * DD * WW + 255) / 256, 256, 0, stream>>>(cache, cache_out);
    cache_update_kernel<<<(BB * DD + 255) / 256, 256, 0, stream>>>(
        x, cache, cu, cidx, hinit, cache_out);
}
